// GlobalInfoGenerate_33801392620062
// MI455X (gfx1250) — matrix operand order repaired and checked by reading, not compiled, not run on hardware
//
#include <hip/hip_runtime.h>
#include <hip/hip_bf16.h>

// GCN 3-layer forward for MI455X (gfx1250, wave32).
// GEMM via v_wmma_f32_16x16x32_bf16 with fp32->bf16 hi/lo split (3 WMMAs per
// tile => ~fp32 accuracy at bf16 matrix-pipe rate). Edge aggregation via
// global f32 atomics into an L2-resident (51.2 MB < 192 MB) accumulator.

typedef __attribute__((ext_vector_type(16))) __bf16 v16bf;
typedef __attribute__((ext_vector_type(8)))  float  v8f;

#define EMB 128
#define BN_EPS 1e-5f

// ---------------------------------------------------------------------------
// One-time graph preprocessing: deg (in-degree of col), dinv, per-edge norm.
// ---------------------------------------------------------------------------
__global__ void deg_kernel(const int* __restrict__ col, float* __restrict__ deg, int E) {
  int e = blockIdx.x * 256 + threadIdx.x;
  if (e < E) unsafeAtomicAdd(&deg[col[e]], 1.0f);
}

__global__ void dinv_kernel(float* __restrict__ deg, int N) {
  int i = blockIdx.x * 256 + threadIdx.x;
  if (i < N) {
    float d = deg[i];
    deg[i] = (d > 0.0f) ? rsqrtf(d) : 0.0f;  // in-place: deg -> dinv
  }
}

__global__ void norm_kernel(const int* __restrict__ row, const int* __restrict__ col,
                            const float* __restrict__ dinv, float* __restrict__ nrm, int E) {
  int e = blockIdx.x * 256 + threadIdx.x;
  if (e < E) nrm[e] = dinv[row[e]] * dinv[col[e]];
}

// ---------------------------------------------------------------------------
// Pack W_l (128x128 fp32, row-major [k][n]) into the per-lane B-fragment
// layout of V_WMMA_F32_16X16X32_BF16, split into bf16 hi/lo parts.
// Storage: [kc(4)][nt(8)][lane(32)][elem(16)] contiguous -> one 32B load/lane.
// B 32x16 layout (matches A): lane L (kh = L>>4) holds, in VGPR g,
// K = (g<4?0:16) + (g&3)*2 + {0,1} + kh*8; N = lane & 15.
// ---------------------------------------------------------------------------
__global__ void pack_w_kernel(const float* __restrict__ Wl,
                              __bf16* __restrict__ whi, __bf16* __restrict__ wlo) {
  int t = blockIdx.x * 256 + threadIdx.x;          // 0 .. 16383
  if (t >= EMB * EMB) return;
  int e    = t & 15;
  int lane = (t >> 4) & 31;
  int nt   = (t >> 9) & 7;
  int kc   = t >> 12;
  int g = e >> 1, p = e & 1;
  int k = kc * 32 + ((g < 4) ? 0 : 16) + (g & 3) * 2 + p + (lane >> 4) * 8;
  int n = nt * 16 + (lane & 15);
  float v = Wl[k * EMB + n];
  __bf16 hi = (__bf16)v;
  __bf16 lo = (__bf16)(v - (float)hi);
  whi[t] = hi;
  wlo[t] = lo;
}

// ---------------------------------------------------------------------------
// x = h @ W   (M x 128 @ 128 x 128), split-bf16 WMMA.
// Block = 256 threads = 8 waves; each wave computes a 16x128 strip.
// A 16x32 bf16 layout: lane L (m = L&15, kh = L>>4) holds, in VGPR g,
// K = (g<4?0:16) + (g&3)*2 + {0,1} + kh*8.
// ---------------------------------------------------------------------------
__global__ __launch_bounds__(256) void gemm_split_bf16_kernel(
    const float* __restrict__ h, const __bf16* __restrict__ whi,
    const __bf16* __restrict__ wlo, float* __restrict__ x, int nrows) {
  const int lane = threadIdx.x & 31;
  const int wave = threadIdx.x >> 5;
  const int rowBase = blockIdx.x * 128 + wave * 16;
  const int m  = lane & 15;
  const int kh = lane >> 4;

  int row = rowBase + m;
  if (row >= nrows) row = nrows - 1;               // clamp: WMMA needs full EXEC
  const float* __restrict__ hrow = h + (size_t)row * EMB;

  v8f acc[8] = {};

#pragma unroll
  for (int kc = 0; kc < 4; ++kc) {
    // Build A hi/lo fragments (fp32 -> bf16 split, in-register).
    v16bf ahi, alo;
#pragma unroll
    for (int g = 0; g < 8; ++g) {
      int kk = kc * 32 + ((g < 4) ? 0 : 16) + (g & 3) * 2 + kh * 8;
      float2 a = *(const float2*)(hrow + kk);
      __bf16 h0 = (__bf16)a.x;  __bf16 l0 = (__bf16)(a.x - (float)h0);
      __bf16 h1 = (__bf16)a.y;  __bf16 l1 = (__bf16)(a.y - (float)h1);
      ahi[2 * g]     = h0;  ahi[2 * g + 1] = h1;
      alo[2 * g]     = l0;  alo[2 * g + 1] = l1;
    }
#pragma unroll
    for (int nt = 0; nt < 8; ++nt) {
      const v16bf bhi = *(const v16bf*)(whi + ((size_t)(kc * 8 + nt) * 32 + lane) * 16);
      const v16bf blo = *(const v16bf*)(wlo + ((size_t)(kc * 8 + nt) * 32 + lane) * 16);
      // a*b ~= ahi*bhi + ahi*blo + alo*bhi  (drop alo*blo ~ 2^-16 rel.)
      acc[nt] = __builtin_amdgcn_wmma_f32_16x16x32_bf16(false, ahi, false, bhi,
                                                        (short)0, acc[nt], false, false);
      acc[nt] = __builtin_amdgcn_wmma_f32_16x16x32_bf16(false, ahi, false, blo,
                                                        (short)0, acc[nt], false, false);
      acc[nt] = __builtin_amdgcn_wmma_f32_16x16x32_bf16(false, alo, false, bhi,
                                                        (short)0, acc[nt], false, false);
    }
  }

  // C/D layout: VGPR r holds (M = r + kh*8, N = m).
#pragma unroll
  for (int nt = 0; nt < 8; ++nt) {
#pragma unroll
    for (int r = 0; r < 8; ++r) {
      int orow = rowBase + kh * 8 + r;
      if (orow < nrows) x[(size_t)orow * EMB + nt * 16 + m] = acc[nt][r];
    }
  }
}

// ---------------------------------------------------------------------------
// agg[col[e]] += x[row[e]] * norm[e].  One wave per edge; lane d owns
// features 4d..4d+3 (float4 load, 4 f32 atomics). Both x and agg are L2-hot.
// ---------------------------------------------------------------------------
__global__ __launch_bounds__(256) void scatter_kernel(
    const float* __restrict__ x, const int* __restrict__ row, const int* __restrict__ col,
    const float* __restrict__ nrm, float* __restrict__ agg, int E) {
  int e = blockIdx.x * 8 + (threadIdx.x >> 5);
  if (e >= E) return;
  int lane = threadIdx.x & 31;
  int r = row[e], c = col[e];
  float w = nrm[e];
  const float4 v = *(const float4*)(x + (size_t)r * EMB + lane * 4);
  float* dst = agg + (size_t)c * EMB + lane * 4;
  unsafeAtomicAdd(dst + 0, v.x * w);
  unsafeAtomicAdd(dst + 1, v.y * w);
  unsafeAtomicAdd(dst + 2, v.z * w);
  unsafeAtomicAdd(dst + 3, v.w * w);
}

// ---------------------------------------------------------------------------
// BatchNorm: per-feature sum / sum-of-squares (block-reduced, then atomics).
// stats[0..127] = sum, stats[128..255] = sumsq.
// ---------------------------------------------------------------------------
__global__ __launch_bounds__(256) void bn_stats_kernel(
    const float* __restrict__ agg, float* __restrict__ stats, int nrows) {
  __shared__ float s0[256], s1[256];
  int d = threadIdx.x & 127, half = threadIdx.x >> 7;
  float sum = 0.0f, sq = 0.0f;
  for (int r = blockIdx.x * 2 + half; r < nrows; r += gridDim.x * 2) {
    float v = agg[(size_t)r * EMB + d];
    sum += v;
    sq  += v * v;
  }
  s0[threadIdx.x] = sum;
  s1[threadIdx.x] = sq;
  __syncthreads();
  if (half == 0) {
    unsafeAtomicAdd(&stats[d],       s0[threadIdx.x] + s0[threadIdx.x + 128]);
    unsafeAtomicAdd(&stats[128 + d], s1[threadIdx.x] + s1[threadIdx.x + 128]);
  }
}

// st[0..127] = scale = gamma * rsqrt(var+eps); st[128..255] = shift.
// Note: conv bias b cancels inside BN ((h+b) - mean(h+b) == h - mean(h)),
// so it is deliberately omitted.
__global__ void bn_finalize_kernel(const float* __restrict__ stats,
                                   const float* __restrict__ gamma,
                                   const float* __restrict__ beta,
                                   float* __restrict__ st, float invN) {
  int d = threadIdx.x;  // 128 threads
  float mu  = stats[d] * invN;
  float var = stats[128 + d] * invN - mu * mu;
  float s = rsqrtf(var + BN_EPS) * gamma[d];
  st[d]       = s;
  st[128 + d] = beta[d] - mu * s;
}

__global__ __launch_bounds__(256) void bn_apply_kernel(
    const float* __restrict__ agg, const float* __restrict__ st,
    float* __restrict__ out, int nquads) {
  int i = blockIdx.x * 256 + threadIdx.x;
  if (i >= nquads) return;
  int d = (i & 31) * 4;
  float4 v = ((const float4*)agg)[i];
  float4 o;
  o.x = v.x * st[d + 0] + st[128 + d + 0];
  o.y = v.y * st[d + 1] + st[128 + d + 1];
  o.z = v.z * st[d + 2] + st[128 + d + 2];
  o.w = v.w * st[d + 3] + st[128 + d + 3];
  ((float4*)out)[i] = o;
}

// ---------------------------------------------------------------------------
extern "C" void kernel_launch(void* const* d_in, const int* in_sizes, int n_in,
                              void* d_out, int out_size, void* d_ws, size_t ws_size,
                              hipStream_t stream) {
  const float* feats = (const float*)d_in[0];
  const int*   eidx  = (const int*)d_in[1];   // [2][E]: row then col (int32)
  const float* W     = (const float*)d_in[2]; // [3][128][128]
  // d_in[3] = b (cancels in BN), d_in[4] = gamma, d_in[5] = beta
  const float* gamma = (const float*)d_in[4];
  const float* beta  = (const float*)d_in[5];
  float* out = (float*)d_out;

  const int N = in_sizes[0] / EMB;     // 100000
  const int E = in_sizes[1] / 2;       // 1600000
  const int* erow = eidx;
  const int* ecol = eidx + E;

  // ---- workspace carve-up (256B aligned) ----
  auto align256 = [](size_t v) { return (v + 255) & ~(size_t)255; };
  char* w = (char*)d_ws;
  size_t featBytes = align256((size_t)N * EMB * sizeof(float));
  float* xbuf  = (float*)w;                     w += featBytes;         // GEMM out
  float* agg   = (float*)w;                     w += featBytes;         // scatter out
  float* nrm   = (float*)w;                     w += align256((size_t)E * sizeof(float));
  float* dinv  = (float*)w;                     w += align256((size_t)N * sizeof(float));
  float* stats = (float*)w;                     w += align256(256 * sizeof(float));
  float* st    = (float*)w;                     w += align256(256 * sizeof(float));
  __bf16* whi  = (__bf16*)w;                    w += align256((size_t)EMB * EMB * sizeof(__bf16));
  __bf16* wlo  = (__bf16*)w;                    w += align256((size_t)EMB * EMB * sizeof(__bf16));
  if ((size_t)(w - (char*)d_ws) > ws_size) return;  // insufficient scratch

  const int eBlocks    = (E + 255) / 256;
  const int nBlocks    = (N + 255) / 256;
  const int gemmBlocks = (N + 127) / 128;          // 8 waves x 16 rows
  const int scatBlocks = (E + 7) / 8;              // 1 wave per edge
  const int quads      = N * EMB / 4;
  const int applyBlks  = (quads + 255) / 256;

  // ---- one-time graph preprocessing ----
  hipMemsetAsync(dinv, 0, (size_t)N * sizeof(float), stream);
  deg_kernel<<<eBlocks, 256, 0, stream>>>(ecol, dinv, E);
  dinv_kernel<<<nBlocks, 256, 0, stream>>>(dinv, N);
  norm_kernel<<<eBlocks, 256, 0, stream>>>(erow, ecol, dinv, nrm, E);

  const float* hcur = feats;
  for (int l = 0; l < 3; ++l) {
    pack_w_kernel<<<64, 256, 0, stream>>>(W + (size_t)l * EMB * EMB, whi, wlo);
    hipMemsetAsync(agg, 0, (size_t)N * EMB * sizeof(float), stream);
    gemm_split_bf16_kernel<<<gemmBlocks, 256, 0, stream>>>(hcur, whi, wlo, xbuf, N);
    scatter_kernel<<<scatBlocks, 256, 0, stream>>>(xbuf, erow, ecol, nrm, agg, E);
    hipMemsetAsync(stats, 0, 256 * sizeof(float), stream);
    bn_stats_kernel<<<256, 256, 0, stream>>>(agg, stats, N);
    bn_finalize_kernel<<<1, 128, 0, stream>>>(stats, gamma + l * EMB, beta + l * EMB,
                                              st, 1.0f / (float)N);
    // d_out doubles as the inter-layer h buffer: its previous contents were
    // fully consumed by this layer's GEMM before this write.
    bn_apply_kernel<<<applyBlks, 256, 0, stream>>>(agg, st, out, quads);
    hcur = out;
  }
}
